// Mlp_moe_2886218023215
// MI455X (gfx1250) — compile-verified, matrix operand order checked by reading, not run on hardware
//
#include <hip/hip_runtime.h>
#include <math.h>

#define NCLS   6
#define NATOMS 5
#define BDIM   768
#define HDIM   3072
#define BATCH  64
#define PTOK   196
#define SEQ    (NCLS + PTOK)        // 202
#define MPATCH (BATCH * PTOK)       // 12544
#define MCLS   (BATCH * NCLS)       // 384

typedef __attribute__((ext_vector_type(16))) __bf16 v16bf;
typedef __attribute__((ext_vector_type(8)))  float  v8f;
typedef __attribute__((ext_vector_type(4)))  int    v4i;

#define TK   32       // K step per WMMA pass
#define TBLK 128      // block tile (M and N)
#define LDSW 48       // shorts per LDS row: 32 data + 16 pad -> 96B rows, 16B aligned

// ---- async global->LDS support (probed; falls back to plain copies) -------
#if defined(__has_builtin)
#if __has_builtin(__builtin_amdgcn_global_load_async_to_lds_b128)
#define HAVE_ASYNC_LDS 1
#endif
#endif
#ifndef HAVE_ASYNC_LDS
#define HAVE_ASYNC_LDS 0
#endif

#if HAVE_ASYNC_LDS
typedef __attribute__((address_space(1))) v4i* gv4i_p;   // global (AS1) int4*
typedef __attribute__((address_space(3))) v4i* sv4i_p;   // LDS    (AS3) int4*
#endif

__device__ __forceinline__ void copy16(const unsigned short* g, unsigned short* l) {
#if HAVE_ASYNC_LDS
    __builtin_amdgcn_global_load_async_to_lds_b128(
        (gv4i_p)(v4i*)g, (sv4i_p)(v4i*)l, 0, 0);
#else
    *(uint4*)l = *(const uint4*)g;
#endif
}

template <int N>
__device__ __forceinline__ void wait_async() {
#if HAVE_ASYNC_LDS
#if __has_builtin(__builtin_amdgcn_s_wait_asynccnt)
    __builtin_amdgcn_s_wait_asynccnt(N);
#else
    asm volatile("s_wait_asynccnt %0" :: "i"(N) : "memory");
#endif
#endif
}

// ---- helpers --------------------------------------------------------------
// Affine row map: r -> (r/period)*stride + (r%period) + off. period=1<<30 => identity.
__device__ __forceinline__ int maprow(int r, int period, int stride, int off) {
    return (r / period) * stride + (r % period) + off;
}

// fp32 -> (hi, lo) bf16 pair, both RNE. hi + lo ~= f with ~16 mantissa bits.
__device__ __forceinline__ void split_bf16(float f, unsigned short& hi, unsigned short& lo) {
    unsigned int u  = __float_as_uint(f);
    unsigned int rh = u + 0x7fffu + ((u >> 16) & 1u);
    hi = (unsigned short)(rh >> 16);
    float fh = __uint_as_float(((unsigned int)hi) << 16);
    float fr = f - fh;
    unsigned int ul = __float_as_uint(fr);
    unsigned int rl = ul + 0x7fffu + ((ul >> 16) & 1u);
    lo = (unsigned short)(rl >> 16);
}

union FragU { v16bf v; unsigned int u[8]; };

// 16x32 bf16 fragment from LDS tile [row][K] (LDSW stride). ISA 7.12.2 layout:
// lanes 0-15 hold K = {0..7,16..23}; lanes 16-31 hold K = {8..15,24..31}.
__device__ __forceinline__ v16bf load_frag(const unsigned short* srow, int half) {
    FragU f;
    const unsigned int* p0 = (const unsigned int*)(srow + half * 8);
    const unsigned int* p1 = (const unsigned int*)(srow + half * 8 + 16);
    f.u[0] = p0[0]; f.u[1] = p0[1]; f.u[2] = p0[2]; f.u[3] = p0[3];
    f.u[4] = p1[0]; f.u[5] = p1[1]; f.u[6] = p1[2]; f.u[7] = p1[3];
    return f.v;
}

// ---- main GEMM: C = maybe_gelu(A.Bt + bias), A/B pre-split bf16 hi/lo planes.
// 3 WMMAs per tile: Ahi*Bhi + Ahi*Blo + Alo*Bhi, f32 accumulate.
// Double-buffered LDS; staging is pure 16B copies (async-to-LDS when available).
// Requires M%128==0, N%128==0, K%32==0 (true for all calls here).
__global__ __launch_bounds__(256) void gemm_pair(
    const unsigned short* __restrict__ Ahi, const unsigned short* __restrict__ Alo,
    int lda, int aperiod, int astride, int aoff,
    const unsigned short* __restrict__ Bhi, const unsigned short* __restrict__ Blo, int ldb,
    const float* __restrict__ bias,
    float* __restrict__ C,
    unsigned short* __restrict__ Chi, unsigned short* __restrict__ Clo,
    int ldc, int operiod, int ostride, int ooff,
    int K, int do_gelu, int split_out)
{
    __shared__ unsigned short sA[2][2][TBLK][LDSW];   // [buf][hi/lo][row][k]
    __shared__ unsigned short sB[2][2][TBLK][LDSW];

    const int tid  = threadIdx.x;
    const int lane = tid & 31;
    const int wave = tid >> 5;
    const int wm   = (wave & 3) << 5;    // wave M offset: 0/32/64/96
    const int wn   = (wave >> 2) << 6;   // wave N offset: 0/64
    const int bm   = blockIdx.y * TBLK;
    const int bn   = blockIdx.x * TBLK;
    const int half = lane >> 4;
    const int l15  = lane & 15;

    v8f acc[2][4];
    for (int mi = 0; mi < 2; ++mi)
        for (int ni = 0; ni < 4; ++ni)
            for (int e = 0; e < 8; ++e)
                acc[mi][ni][e] = 0.0f;

    // Stage one 128x32 K-tile of all four planes into LDS buffer `buf`.
    // 512 chunks of 16B per plane; 256 threads x 2 chunks x 4 planes.
    auto stage = [&](int buf, int k0) {
        for (int i = 0; i < 2; ++i) {
            int c   = tid + (i << 8);        // 0..511
            int row = c >> 2;                // 0..127
            int ko  = (c & 3) << 3;          // 0,8,16,24 (shorts)
            int gra = maprow(bm + row, aperiod, astride, aoff);
            size_t ae = (size_t)gra * lda + (k0 + ko);
            size_t be = (size_t)(bn + row) * ldb + (k0 + ko);
            copy16(Ahi + ae, &sA[buf][0][row][ko]);
            copy16(Alo + ae, &sA[buf][1][row][ko]);
            copy16(Bhi + be, &sB[buf][0][row][ko]);
            copy16(Blo + be, &sB[buf][1][row][ko]);
        }
    };

    stage(0, 0);
    int buf = 0;
    for (int k0 = 0; k0 < K; k0 += TK) {
        const bool more = (k0 + TK) < K;
        if (more) {
            stage(buf ^ 1, k0 + TK);   // prefetch next tile into other buffer
            wait_async<8>();           // retire current tile's batch (issued earlier)
        } else {
            wait_async<0>();
        }
        __syncthreads();

        v16bf ah[2], al[2], bh[4], bl[4];
        for (int mi = 0; mi < 2; ++mi) {
            int r = wm + (mi << 4) + l15;
            ah[mi] = load_frag(sA[buf][0][r], half);
            al[mi] = load_frag(sA[buf][1][r], half);
        }
        for (int ni = 0; ni < 4; ++ni) {
            int r = wn + (ni << 4) + l15;
            bh[ni] = load_frag(sB[buf][0][r], half);
            bl[ni] = load_frag(sB[buf][1][r], half);
        }
        for (int mi = 0; mi < 2; ++mi)
            for (int ni = 0; ni < 4; ++ni) {
                acc[mi][ni] = __builtin_amdgcn_wmma_f32_16x16x32_bf16(
                    false, ah[mi], false, bh[ni], (short)0, acc[mi][ni], false, false);
                acc[mi][ni] = __builtin_amdgcn_wmma_f32_16x16x32_bf16(
                    false, ah[mi], false, bl[ni], (short)0, acc[mi][ni], false, false);
                acc[mi][ni] = __builtin_amdgcn_wmma_f32_16x16x32_bf16(
                    false, al[mi], false, bh[ni], (short)0, acc[mi][ni], false, false);
            }
        __syncthreads();
        buf ^= 1;
    }

    // Epilogue: lane = (m>=8)*16 + n, vgpr e -> m = e + 8*half.
    for (int mi = 0; mi < 2; ++mi)
        for (int ni = 0; ni < 4; ++ni) {
            int col = bn + wn + (ni << 4) + l15;
            float bv = bias[col];
            for (int e = 0; e < 8; ++e) {
                int r    = bm + wm + (mi << 4) + (half << 3) + e;
                int orow = maprow(r, operiod, ostride, ooff);
                float v  = acc[mi][ni][e] + bv;
                if (do_gelu) v = 0.5f * v * (1.0f + erff(v * 0.7071067811865475f));
                size_t idx = (size_t)orow * ldc + col;
                if (split_out) {
                    unsigned short h, l;
                    split_bf16(v, h, l);
                    Chi[idx] = h; Clo[idx] = l;
                } else {
                    C[idx] = v;
                }
            }
        }
}

// fp32 array -> separate bf16 hi/lo planes (n must be a multiple of 4).
__global__ void split_planes(const float* __restrict__ src,
                             unsigned short* __restrict__ hi,
                             unsigned short* __restrict__ lo, int n4)
{
    int i = blockIdx.x * 256 + threadIdx.x;
    if (i >= n4) return;
    float4 f = ((const float4*)src)[i];
    unsigned short h0, l0, h1, l1, h2, l2, h3, l3;
    split_bf16(f.x, h0, l0);
    split_bf16(f.y, h1, l1);
    split_bf16(f.z, h2, l2);
    split_bf16(f.w, h3, l3);
    ((ushort4*)hi)[i] = make_ushort4(h0, h1, h2, h3);
    ((ushort4*)lo)[i] = make_ushort4(l0, l1, l2, l3);
}

// One wave per (b, n): gate logit, sigmoid weight, src/dst atom selection.
__global__ void gate_kernel(const float* __restrict__ x,
                            const float* __restrict__ gate_delta,
                            float* __restrict__ wv,
                            int* __restrict__ srcv,
                            int* __restrict__ dstv)
{
    const int r = blockIdx.x;                // 0..383
    const int b = r / NCLS, n = r % NCLS;
    const float* xv = x + ((size_t)b * SEQ + n) * BDIM;
    const float* gd = gate_delta + (size_t)n * BDIM;
    float s = 0.0f;
    for (int k = threadIdx.x; k < BDIM; k += 32) s += xv[k] * gd[k];
    for (int off = 16; off > 0; off >>= 1) s += __shfl_down(s, off, 32);
    if (threadIdx.x == 0) {
        const int LEFTK [NCLS] = { 3,  4,  8,  9, 13, 14};
        const int RIGHTK[NCLS] = {15, 20, 16, 21, 17, 22};
        bool  left = (s >= 0.0f);
        float p    = 1.0f / (1.0f + expf(-s));
        int   key  = left ? LEFTK[n] : RIGHTK[n];
        wv[r]   = left ? p : 1.0f - p;
        srcv[r] = key / NATOMS;
        dstv[r] = key % NATOMS;
    }
}

// hid[r,h] = hid_all[r, src[r]*HDIM + h], emitted as split bf16 planes.
__global__ void gather_split_hid(const float* __restrict__ hid_all,
                                 const int* __restrict__ srcv,
                                 unsigned short* __restrict__ hhi,
                                 unsigned short* __restrict__ hlo)
{
    int idx = blockIdx.x * 256 + threadIdx.x;     // < 384*3072
    int r = idx / HDIM;
    int h = idx - r * HDIM;
    float v = hid_all[(size_t)r * (NATOMS * HDIM) + (size_t)srcv[r] * HDIM + h];
    unsigned short vh, vl;
    split_bf16(v, vh, vl);
    hhi[idx] = vh; hlo[idx] = vl;
}

// out[b*SEQ + n, d] = out_all[r, dst[r]*BDIM + d] * w[r]
__global__ void cls_finish(const float* __restrict__ out_all,
                           const int* __restrict__ dstv,
                           const float* __restrict__ wv,
                           float* __restrict__ out)
{
    int idx = blockIdx.x * 256 + threadIdx.x;     // < 384*768
    int r = idx / BDIM;
    int d = idx - r * BDIM;
    int b = r / NCLS, n = r % NCLS;
    float v = out_all[(size_t)r * (NATOMS * BDIM) + (size_t)dstv[r] * BDIM + d] * wv[r];
    out[((size_t)b * SEQ + n) * BDIM + d] = v;
}

extern "C" void kernel_launch(void* const* d_in, const int* in_sizes, int n_in,
                              void* d_out, int out_size, void* d_ws, size_t ws_size,
                              hipStream_t stream)
{
    (void)in_sizes; (void)n_in; (void)out_size; (void)ws_size;
    const float* x      = (const float*)d_in[0];
    const float* pw1    = (const float*)d_in[1];
    const float* pb1    = (const float*)d_in[2];
    const float* pw2    = (const float*)d_in[3];
    const float* pb2    = (const float*)d_in[4];
    const float* gdelta = (const float*)d_in[5];
    const float* aiw    = (const float*)d_in[6];   // (15360, 768)
    const float* aib    = (const float*)d_in[7];   // (15360)
    const float* aow    = (const float*)d_in[8];   // (3840, 3072)
    const float* aob    = (const float*)d_in[9];   // (3840)
    float* out = (float*)d_out;                    // (64, 202, 768) fp32

    typedef unsigned short u16;
    char* ws = (char*)d_ws;
    u16*   xhi    = (u16*)  (ws + 0ull);             // 9,922,560 bf16
    u16*   xlo    = (u16*)  (ws + 19845120ull);
    u16*   w1hi   = (u16*)  (ws + 39690240ull);      // 2,359,296
    u16*   w1lo   = (u16*)  (ws + 44408832ull);
    u16*   w2hi   = (u16*)  (ws + 49127424ull);      // 2,359,296
    u16*   w2lo   = (u16*)  (ws + 53846016ull);
    u16*   aiwhi  = (u16*)  (ws + 58564608ull);      // 11,796,480
    u16*   aiwlo  = (u16*)  (ws + 82157568ull);
    u16*   aowhi  = (u16*)  (ws + 105750528ull);     // 11,796,480
    u16*   aowlo  = (u16*)  (ws + 129343488ull);
    u16*   hidhi  = (u16*)  (ws + 152936448ull);     // 12544 x 3072
    u16*   hidlo  = (u16*)  (ws + 230006784ull);
    float* hidall = (float*)(ws + 307077120ull);     // 384 x 15360 fp32
    u16*   hshi   = (u16*)  (ws + 330670080ull);     // 384 x 3072
    u16*   hslo   = (u16*)  (ws + 333029376ull);
    float* outall = (float*)(ws + 335388672ull);     // 384 x 3840 fp32
    float* wv     = (float*)(ws + 341286912ull);
    int*   srcv   = (int*)  (ws + 341288448ull);
    int*   dstv   = (int*)  (ws + 341289984ull);

    const int PLAIN = 1 << 30;

    // Pre-split every GEMM operand into bf16 hi/lo planes (elementwise, once).
    split_planes<<<(9922560 / 4 + 255) / 256, 256, 0, stream>>>(x,   xhi,   xlo,   9922560 / 4);
    split_planes<<<(2359296 / 4 + 255) / 256, 256, 0, stream>>>(pw1, w1hi,  w1lo,  2359296 / 4);
    split_planes<<<(2359296 / 4 + 255) / 256, 256, 0, stream>>>(pw2, w2hi,  w2lo,  2359296 / 4);
    split_planes<<<(11796480 / 4 + 255) / 256, 256, 0, stream>>>(aiw, aiwhi, aiwlo, 11796480 / 4);
    split_planes<<<(11796480 / 4 + 255) / 256, 256, 0, stream>>>(aow, aowhi, aowlo, 11796480 / 4);

    gate_kernel<<<MCLS, 32, 0, stream>>>(x, gdelta, wv, srcv, dstv);

    // hidden = gelu(patch @ W1^T + b1), emitted as split planes.
    gemm_pair<<<dim3(HDIM / 128, MPATCH / 128), 256, 0, stream>>>(
        xhi, xlo, BDIM, PTOK, SEQ, NCLS,
        w1hi, w1lo, BDIM, pb1,
        nullptr, hidhi, hidlo, HDIM, PLAIN, 0, 0,
        BDIM, 1, 1);

    // hid_all = gelu(cls @ atom_in_w^T + atom_in_b), fp32 (gathered next).
    gemm_pair<<<dim3(NATOMS * HDIM / 128, MCLS / 128), 256, 0, stream>>>(
        xhi, xlo, BDIM, NCLS, SEQ, 0,
        aiwhi, aiwlo, BDIM, aib,
        hidall, nullptr, nullptr, NATOMS * HDIM, PLAIN, 0, 0,
        BDIM, 1, 0);

    gather_split_hid<<<(MCLS * HDIM) / 256, 256, 0, stream>>>(hidall, srcv, hshi, hslo);

    // patch_out = hidden @ W2^T + b2 -> d_out patch rows.
    gemm_pair<<<dim3(BDIM / 128, MPATCH / 128), 256, 0, stream>>>(
        hidhi, hidlo, HDIM, PLAIN, 0, 0,
        w2hi, w2lo, HDIM, pb2,
        out, nullptr, nullptr, BDIM, PTOK, SEQ, NCLS,
        HDIM, 0, 0);

    // out_all = hid @ atom_out_w^T + atom_out_b, fp32.
    gemm_pair<<<dim3(NATOMS * BDIM / 128, MCLS / 128), 256, 0, stream>>>(
        hshi, hslo, HDIM, PLAIN, 0, 0,
        aowhi, aowlo, HDIM, aob,
        outall, nullptr, nullptr, NATOMS * BDIM, PLAIN, 0, 0,
        HDIM, 0, 0);

    cls_finish<<<(MCLS * BDIM) / 256, 256, 0, stream>>>(outall, dstv, wv, out);
}
